// SelfAttention_48627619725549
// MI455X (gfx1250) — matrix operand order repaired and checked by reading, not compiled, not run on hardware
//
#include <hip/hip_runtime.h>
#include <hip/hip_bf16.h>

typedef __bf16 bf16;
typedef __attribute__((ext_vector_type(16))) __bf16 v16bf;
typedef __attribute__((ext_vector_type(8)))  __bf16 v8bf;
typedef __attribute__((ext_vector_type(8)))  float   v8f;

constexpr int BB = 4;
constexpr int TT = 2048;
constexpr int CC = 1024;
constexpr int HD = 64;

__device__ __forceinline__ v8f wmma_bf16(v16bf a, v16bf b, v8f c) {
    // 8 args: (neg_a, A, neg_b, B, c_mod, C, reuse_a, reuse_b)
    return __builtin_amdgcn_wmma_f32_16x16x32_bf16(false, a, false, b, (short)0, c, false, false);
}

// Build a v16bf A-fragment from two contiguous 8-element (16B) chunks.
template <typename PTR>
__device__ __forceinline__ v16bf load2x8(PTR p0, PTR p1) {
    v8bf lo = *reinterpret_cast<const v8bf*>(p0);
    v8bf hi = *reinterpret_cast<const v8bf*>(p1);
    v16bf r;
#pragma unroll
    for (int i = 0; i < 8; ++i) { r[i] = lo[i]; r[i + 8] = hi[i]; }
    return r;
}

// ---------------------------------------------------------------------------
// Kernel 0: transpose+convert weights [C,HD] f32 -> [HD,C] bf16 (B-frag friendly)
// ---------------------------------------------------------------------------
__global__ __launch_bounds__(256) void wconv_kernel(
    const float* __restrict__ Wq, const float* __restrict__ Wk, const float* __restrict__ Wv,
    bf16* __restrict__ wtq, bf16* __restrict__ wtk, bf16* __restrict__ wtv)
{
    int idx = blockIdx.x * 256 + threadIdx.x;   // 0 .. HD*CC-1
    int hd  = idx >> 10;                        // / CC
    int c   = idx & (CC - 1);
    wtq[idx] = (bf16)Wq[c * HD + hd];
    wtk[idx] = (bf16)Wk[c * HD + hd];
    wtv[idx] = (bf16)Wv[c * HD + hd];
}

// ---------------------------------------------------------------------------
// Kernel 1: fused QKV projection. One block = one 16-row tile of x.
// 4 waves; wave w computes output cols [16w,16w+16) of Q, K, V via bf16 WMMA.
// Writes Q,K row-major [B*T][HD] bf16 and V transposed [B][HD][T] bf16.
// ---------------------------------------------------------------------------
__global__ __launch_bounds__(128) void proj_kernel(
    const float* __restrict__ x,
    const bf16* __restrict__ wtq, const bf16* __restrict__ wtk, const bf16* __restrict__ wtv,
    bf16* __restrict__ Q, bf16* __restrict__ K, bf16* __restrict__ Vt)
{
    __shared__ bf16 xs[16][1032];               // 16 x CC bf16, padded row stride (16B mult)

    const int tile = blockIdx.x;                // 0 .. B*T/16-1
    const int b    = tile / (TT / 16);
    const int t0   = (tile % (TT / 16)) * 16;
    const int tid  = threadIdx.x;

    // Cooperative load of the x tile (16 x 1024 f32), convert to bf16 in LDS.
    const float* xg = x + (size_t)(b * TT + t0) * CC;
#pragma unroll 4
    for (int it = 0; it < 32; ++it) {
        int idx = it * 128 + tid;               // float4 index 0..4095
        int row = idx >> 8;                     // / 256
        int c4  = idx & 255;
        float4 f = reinterpret_cast<const float4*>(xg + (size_t)row * CC)[c4];
        bf16* d = &xs[row][c4 * 4];
        d[0] = (bf16)f.x; d[1] = (bf16)f.y; d[2] = (bf16)f.z; d[3] = (bf16)f.w;
    }
    __syncthreads();

    const int wave  = tid >> 5;                 // n-tile index (0..3)
    const int lane  = tid & 31;
    const int ln    = lane & 15;
    const int hi    = lane >> 4;
    const int abase = hi * 16;                  // A-frag K sub-offset
    const int bbase = hi * 16;                  // B-frag K sub-offset

    v8f accq = {}, acck = {}, accv = {};
    const bf16* wq = wtq + (size_t)(wave * 16 + ln) * CC;
    const bf16* wk = wtk + (size_t)(wave * 16 + ln) * CC;
    const bf16* wv = wtv + (size_t)(wave * 16 + ln) * CC;

    for (int kk = 0; kk < CC / 32; ++kk) {
        const int k0 = kk * 32;
        v16bf a  = load2x8(&xs[ln][k0 + abase], &xs[ln][k0 + 8 + abase]);
        v16bf bq = *reinterpret_cast<const v16bf*>(wq + k0 + bbase);
        v16bf bk = *reinterpret_cast<const v16bf*>(wk + k0 + bbase);
        v16bf bv = *reinterpret_cast<const v16bf*>(wv + k0 + bbase);
        accq = wmma_bf16(a, bq, accq);
        acck = wmma_bf16(a, bk, acck);
        accv = wmma_bf16(a, bv, accv);
    }

    const int hd = wave * 16 + ln;
#pragma unroll
    for (int r = 0; r < 8; ++r) {
        int t = t0 + r + hi * 8;
        Q[(size_t)(b * TT + t) * HD + hd] = (bf16)accq[r];
        K[(size_t)(b * TT + t) * HD + hd] = (bf16)acck[r];
    }
    {   // V transposed: contiguous 8 bf16 along t
        v8bf vv;
#pragma unroll
        for (int r = 0; r < 8; ++r) vv[r] = (bf16)accv[r];
        *reinterpret_cast<v8bf*>(&Vt[((size_t)(b * HD + hd)) * TT + t0 + hi * 8]) = vv;
    }
}

// ---------------------------------------------------------------------------
// Kernel 2: double-softmax flash attention. One wave = one 16-query tile.
// Pass 1: per-lane online (m, l) over all T keys, merged across lanes ONCE
//         at the end (two-level online softmax; no shuffles in the hot loop).
// Pass 2: recompute scores causally, e2 = exp(exp(s-m)/l), accumulate
//         O = sum(e2 * v) and sum2 = sum(e2); out = O / sum2.
// ---------------------------------------------------------------------------
__global__ __launch_bounds__(128) void attn_kernel(
    const bf16* __restrict__ Q, const bf16* __restrict__ Kk, const bf16* __restrict__ Vt,
    float* __restrict__ out)
{
    __shared__ bf16 P[4][16][48];               // per-wave 16x32 P tile, padded stride

    const int wave  = threadIdx.x >> 5;
    const int lane  = threadIdx.x & 31;
    const int ln    = lane & 15;
    const int hi    = lane >> 4;
    const int abase = hi * 16;
    const int bbase = hi * 16;

    const int tile = blockIdx.x * 4 + wave;     // 0 .. B*T/16-1
    const int b    = tile / (TT / 16);
    const int i0   = (tile % (TT / 16)) * 16;

    const bf16* qb = Q  + (size_t)(b * TT) * HD;
    const bf16* kb = Kk + (size_t)(b * TT) * HD;
    const bf16* vb = Vt + (size_t)(b * HD) * TT;

    // Q A-fragments: K-dim = HD split into [0,32) and [32,64)
    const bf16* qrow = qb + (size_t)(i0 + ln) * HD;
    v16bf aq0 = load2x8(qrow + abase,      qrow + 8 + abase);
    v16bf aq1 = load2x8(qrow + 32 + abase, qrow + 40 + abase);

    const float scale = 8.0f;                   // sqrt(HD), multiplied (faithful)

    // ---- pass 1: per-lane online softmax stats over all keys -----------------
    float m[8], l[8];
#pragma unroll
    for (int r = 0; r < 8; ++r) { m[r] = -3.0e38f; l[r] = 0.0f; }

    for (int j0 = 0; j0 < TT; j0 += 32) {
        v8f sA = {}, sB = {};
        {
            const bf16* krow = kb + (size_t)(j0 + ln) * HD;
            v16bf b0 = *reinterpret_cast<const v16bf*>(krow + bbase);
            v16bf b1 = *reinterpret_cast<const v16bf*>(krow + 32 + bbase);
            sA = wmma_bf16(aq0, b0, sA);
            sA = wmma_bf16(aq1, b1, sA);
        }
        {
            const bf16* krow = kb + (size_t)(j0 + 16 + ln) * HD;
            v16bf b0 = *reinterpret_cast<const v16bf*>(krow + bbase);
            v16bf b1 = *reinterpret_cast<const v16bf*>(krow + 32 + bbase);
            sB = wmma_bf16(aq0, b0, sB);
            sB = wmma_bf16(aq1, b1, sB);
        }
#pragma unroll
        for (int r = 0; r < 8; ++r) {
            float xa = sA[r] * scale;
            float xb = sB[r] * scale;
            float mn = fmaxf(m[r], fmaxf(xa, xb));
            l[r] = l[r] * __expf(m[r] - mn) + __expf(xa - mn) + __expf(xb - mn);
            m[r] = mn;
        }
    }
    // Merge per-lane (m, l) across the 16-lane half holding each row.
#pragma unroll
    for (int r = 0; r < 8; ++r) {
        float mr = m[r], lr = l[r];
#pragma unroll
        for (int mask = 1; mask <= 8; mask <<= 1) {
            float om = __shfl_xor(mr, mask);
            float ol = __shfl_xor(lr, mask);
            float mn = fmaxf(mr, om);
            lr = lr * __expf(mr - mn) + ol * __expf(om - mn);
            mr = mn;
        }
        m[r] = mr;
        l[r] = 1.0f / lr;                       // reciprocal for pass 2
    }

    // ---- pass 2: causal recompute + second softmax + PV ----------------------
    v8f o0 = {}, o1 = {}, o2 = {}, o3 = {};
    float s2[8];
#pragma unroll
    for (int r = 0; r < 8; ++r) s2[r] = 0.0f;

    bf16* Pw = &P[wave][0][0];
    const int nchunks = (i0 + 16 + 31) / 32;    // 32-key chunks covering causal range
    for (int ch = 0; ch < nchunks; ++ch) {
        const int j0 = ch * 32;
        v8f sA = {}, sB = {};
        {
            const bf16* krow = kb + (size_t)(j0 + ln) * HD;
            v16bf b0 = *reinterpret_cast<const v16bf*>(krow + bbase);
            v16bf b1 = *reinterpret_cast<const v16bf*>(krow + 32 + bbase);
            sA = wmma_bf16(aq0, b0, sA);
            sA = wmma_bf16(aq1, b1, sA);
        }
        {
            const bf16* krow = kb + (size_t)(j0 + 16 + ln) * HD;
            v16bf b0 = *reinterpret_cast<const v16bf*>(krow + bbase);
            v16bf b1 = *reinterpret_cast<const v16bf*>(krow + 32 + bbase);
            sB = wmma_bf16(aq0, b0, sB);
            sB = wmma_bf16(aq1, b1, sB);
        }
#pragma unroll
        for (int r = 0; r < 8; ++r) {
            int i   = i0 + r + hi * 8;          // query row of this slot
            int row = r + hi * 8;
            float eA = 0.0f, eB = 0.0f;
            if (j0 + ln <= i)      eA = __expf(__expf(sA[r] * scale - m[r]) * l[r]);
            if (j0 + 16 + ln <= i) eB = __expf(__expf(sB[r] * scale - m[r]) * l[r]);
            s2[r] += eA + eB;
            Pw[row * 48 + ln]      = (bf16)eA;  // C/D layout -> LDS (16x32 tile)
            Pw[row * 48 + 16 + ln] = (bf16)eB;
        }
        asm volatile("s_wait_dscnt 0" ::: "memory");   // LDS writes visible before re-read
        // Re-read P as a 16x32 A-fragment
        v16bf ap = load2x8(&Pw[ln * 48 + abase], &Pw[ln * 48 + 8 + abase]);
        // V B-fragments (Vt rows are contiguous over t) and accumulate
        v16bf bv0 = *reinterpret_cast<const v16bf*>(vb + (size_t)(ln)      * TT + j0 + bbase);
        v16bf bv1 = *reinterpret_cast<const v16bf*>(vb + (size_t)(16 + ln) * TT + j0 + bbase);
        v16bf bv2 = *reinterpret_cast<const v16bf*>(vb + (size_t)(32 + ln) * TT + j0 + bbase);
        v16bf bv3 = *reinterpret_cast<const v16bf*>(vb + (size_t)(48 + ln) * TT + j0 + bbase);
        o0 = wmma_bf16(ap, bv0, o0);
        o1 = wmma_bf16(ap, bv1, o1);
        o2 = wmma_bf16(ap, bv2, o2);
        o3 = wmma_bf16(ap, bv3, o3);
    }

    // Row sums of e2 (second softmax denominator), then normalize and store.
#pragma unroll
    for (int r = 0; r < 8; ++r) {
        float v = s2[r];
        v += __shfl_xor(v, 1); v += __shfl_xor(v, 2);
        v += __shfl_xor(v, 4); v += __shfl_xor(v, 8);
        s2[r] = 1.0f / v;
    }
    float* ob = out + (size_t)(b * TT) * HD;
#pragma unroll
    for (int r = 0; r < 8; ++r) {
        int t = i0 + r + hi * 8;
        float inv = s2[r];
        ob[(size_t)t * HD + (ln)]      = o0[r] * inv;
        ob[(size_t)t * HD + (16 + ln)] = o1[r] * inv;
        ob[(size_t)t * HD + (32 + ln)] = o2[r] * inv;
        ob[(size_t)t * HD + (48 + ln)] = o3[r] * inv;
    }
}

// ---------------------------------------------------------------------------
extern "C" void kernel_launch(void* const* d_in, const int* in_sizes, int n_in,
                              void* d_out, int out_size, void* d_ws, size_t ws_size,
                              hipStream_t stream) {
    const float* x  = (const float*)d_in[0];
    const float* Wq = (const float*)d_in[1];
    const float* Wk = (const float*)d_in[2];
    const float* Wv = (const float*)d_in[3];
    float* out = (float*)d_out;

    // Workspace layout (bf16 elements): Wt q/k/v, then Q, K, Vt (~3.4 MB total)
    bf16* wtq = (bf16*)d_ws;
    bf16* wtk = wtq + (size_t)HD * CC;
    bf16* wtv = wtk + (size_t)HD * CC;
    bf16* Qb  = wtv + (size_t)HD * CC;
    bf16* Kb  = Qb  + (size_t)BB * TT * HD;
    bf16* Vtb = Kb  + (size_t)BB * TT * HD;

    wconv_kernel<<<(HD * CC) / 256, 256, 0, stream>>>(Wq, Wk, Wv, wtq, wtk, wtv);
    proj_kernel<<<(BB * TT) / 16, 128, 0, stream>>>(x, wtq, wtk, wtv, Qb, Kb, Vtb);
    attn_kernel<<<(BB * TT) / 64, 128, 0, stream>>>(Qb, Kb, Vtb, out);
}
